// MultiHeadAttention_8546984919667
// MI455X (gfx1250) — compile-verified, matrix operand order checked by reading, not run on hardware
//
#include <hip/hip_runtime.h>
#include <hip/hip_bf16.h>

// ---------------------------------------------------------------------------
// MHA forward for MI455X (gfx1250, wave32, WMMA).
// All GEMMs via v_wmma_f32_16x16x32_bf16 (bf16 inputs, fp32 accumulate).
// Flash-attention streaming: scores never materialized in HBM.
// v2: 128x128 GEMM block tile (8 WMMA/wave/k-step), double-buffered LDS with
//     register prefetch pipeline; K/V register prefetch in attention loop.
// ---------------------------------------------------------------------------

#define D_MODEL 1024
#define SEQ     4096
#define NHEAD   16
#define DK      64

typedef __attribute__((ext_vector_type(16))) __bf16        v16bf;
typedef __attribute__((ext_vector_type(8)))  float         v8f;
typedef __attribute__((ext_vector_type(8)))  unsigned short v8u;
typedef __attribute__((ext_vector_type(4)))  unsigned short v4u;

__device__ __forceinline__ unsigned short f2bf(float f) {
    unsigned int u = __builtin_bit_cast(unsigned int, f);
    u += 0x7FFFu + ((u >> 16) & 1u);       // round-to-nearest-even
    return (unsigned short)(u >> 16);
}

__device__ __forceinline__ v8f zero8() {
    v8f z;
#pragma unroll
    for (int i = 0; i < 8; ++i) z[i] = 0.0f;
    return z;
}

// Load one 16x32 bf16 A/B fragment from LDS laid out as [row][k], row stride
// `stride` halfwords (multiple of 8), k-chunk starting at `koff`.
// ISA layout: lane L holds row (L&15); group g=L>>4; elem i<8 -> K=koff+g*8+i,
// elem i>=8 -> K=koff+16+g*8+(i-8).
__device__ __forceinline__ v16bf load_frag16(const unsigned short* lds,
                                             int rowBase, int stride, int koff) {
    const int lane = threadIdx.x & 31;
    const int row  = rowBase + (lane & 15);
    const int g    = (lane >> 4) & 1;
    const unsigned short* p = lds + row * stride + koff + g * 8;
    v8u lo = *(const v8u*)p;
    v8u hi = *(const v8u*)(p + 16);
    v16bf f;
#pragma unroll
    for (int i = 0; i < 8; ++i) {
        f[i]     = __builtin_bit_cast(__bf16, (unsigned short)lo[i]);
        f[i + 8] = __builtin_bit_cast(__bf16, (unsigned short)hi[i]);
    }
    return f;
}

__device__ __forceinline__ v8f wmma_bf16(v16bf a, v16bf b, v8f c) {
    return __builtin_amdgcn_wmma_f32_16x16x32_bf16(
        /*neg_a=*/false, a, /*neg_b=*/false, b,
        /*c_mod=*/(short)0, c, /*reuse_a=*/false, /*reuse_b=*/false);
}

// ---------------------------------------------------------------------------
// GEMM: C[M=4096, N=1024] = X[4096,1024] * W^T (W is [N,K]) + bias
// MODE 0: X fp32, output bf16 head-major ws[h][s][dk]   (QKV projections)
// MODE 1: X bf16 (workspace), output fp32 row-major d_out (O projection)
// Block tile 128x128, BK=32, 256 threads = 8 waves (4M x 2N), wave tile 32x64.
// Double-buffered LDS, global->register prefetch pipeline.
// ---------------------------------------------------------------------------
#define GEMM_SA 40   // 32 + 8 halfword pad
#define GEMM_SB 40

template <int MODE>
__global__ __launch_bounds__(256)
void gemm_bf16_kernel(const void* __restrict__ Xv,
                      const float* __restrict__ W,
                      const float* __restrict__ bias,
                      void* __restrict__ Outv) {
    __shared__ __align__(16) unsigned short sA[2][128 * GEMM_SA];
    __shared__ __align__(16) unsigned short sB[2][128 * GEMM_SB];

    const int tid = threadIdx.x;
    const int n0  = blockIdx.x * 128;
    const int m0  = blockIdx.y * 128;

    const int w    = tid >> 5;        // wave 0..7
    const int wm   = w >> 1;          // 0..3  (M)
    const int wn   = w & 1;           // 0..1  (N)
    const int mb   = wm * 32;
    const int nb   = wn * 64;
    const int lane = tid & 31;
    const int g    = lane >> 4;
    const int ln   = lane & 15;

    v8f acc[2][4];
#pragma unroll
    for (int i = 0; i < 2; ++i)
#pragma unroll
        for (int j = 0; j < 4; ++j) acc[i][j] = zero8();

    // prefetch registers
    float4 xr[4];   // MODE 0: X fp32 staging (128x32 = 1024 float4 / 256 thr)
    v8u    xr8[2];  // MODE 1: X bf16 staging (128x32 = 512 v8u / 256 thr)
    float4 wr[4];   // W fp32 staging        (128x32 = 1024 float4 / 256 thr)

    auto load_tiles = [&](int k0) {
        if (MODE == 0) {
            const float* X = (const float*)Xv;
#pragma unroll
            for (int t = 0; t < 4; ++t) {
                int f4 = tid + t * 256; int row = f4 >> 3, c4 = f4 & 7;
                xr[t] = *(const float4*)(X + (size_t)(m0 + row) * D_MODEL + k0 + c4 * 4);
            }
        } else {
            const unsigned short* X = (const unsigned short*)Xv;
#pragma unroll
            for (int t = 0; t < 2; ++t) {
                int f8 = tid + t * 256; int row = f8 >> 2, c8 = f8 & 3;
                xr8[t] = *(const v8u*)(X + (size_t)(m0 + row) * D_MODEL + k0 + c8 * 8);
            }
        }
#pragma unroll
        for (int t = 0; t < 4; ++t) {
            int f4 = tid + t * 256; int row = f4 >> 3, c4 = f4 & 7;
            wr[t] = *(const float4*)(W + (size_t)(n0 + row) * D_MODEL + k0 + c4 * 4);
        }
    };

    auto store_tiles = [&](unsigned short* sAb, unsigned short* sBb) {
        if (MODE == 0) {
#pragma unroll
            for (int t = 0; t < 4; ++t) {
                int f4 = tid + t * 256; int row = f4 >> 3, c4 = f4 & 7;
                v4u pk;
                pk[0] = f2bf(xr[t].x); pk[1] = f2bf(xr[t].y);
                pk[2] = f2bf(xr[t].z); pk[3] = f2bf(xr[t].w);
                *(v4u*)(sAb + row * GEMM_SA + c4 * 4) = pk;
            }
        } else {
#pragma unroll
            for (int t = 0; t < 2; ++t) {
                int f8 = tid + t * 256; int row = f8 >> 2, c8 = f8 & 3;
                *(v8u*)(sAb + row * GEMM_SA + c8 * 8) = xr8[t];
            }
        }
#pragma unroll
        for (int t = 0; t < 4; ++t) {
            int f4 = tid + t * 256; int row = f4 >> 3, c4 = f4 & 7;
            v4u pk;
            pk[0] = f2bf(wr[t].x); pk[1] = f2bf(wr[t].y);
            pk[2] = f2bf(wr[t].z); pk[3] = f2bf(wr[t].w);
            *(v4u*)(sBb + row * GEMM_SB + c4 * 4) = pk;
        }
    };

    const int nk = D_MODEL / 32;
    load_tiles(0);
    store_tiles(sA[0], sB[0]);
    __syncthreads();

    for (int kt = 0; kt < nk; ++kt) {
        const int cur = kt & 1;
        if (kt + 1 < nk) load_tiles((kt + 1) * 32);   // overlap with compute

        v16bf a0 = load_frag16(sA[cur], mb,      GEMM_SA, 0);
        v16bf a1 = load_frag16(sA[cur], mb + 16, GEMM_SA, 0);
        v16bf b0 = load_frag16(sB[cur], nb,      GEMM_SB, 0);
        v16bf b1 = load_frag16(sB[cur], nb + 16, GEMM_SB, 0);
        v16bf b2 = load_frag16(sB[cur], nb + 32, GEMM_SB, 0);
        v16bf b3 = load_frag16(sB[cur], nb + 48, GEMM_SB, 0);

        acc[0][0] = wmma_bf16(a0, b0, acc[0][0]);
        acc[0][1] = wmma_bf16(a0, b1, acc[0][1]);
        acc[0][2] = wmma_bf16(a0, b2, acc[0][2]);
        acc[0][3] = wmma_bf16(a0, b3, acc[0][3]);
        acc[1][0] = wmma_bf16(a1, b0, acc[1][0]);
        acc[1][1] = wmma_bf16(a1, b1, acc[1][1]);
        acc[1][2] = wmma_bf16(a1, b2, acc[1][2]);
        acc[1][3] = wmma_bf16(a1, b3, acc[1][3]);

        if (kt + 1 < nk) {
            store_tiles(sA[cur ^ 1], sB[cur ^ 1]);
            __syncthreads();
        }
    }

    // --- epilogue ---
#pragma unroll
    for (int i = 0; i < 2; ++i)
#pragma unroll
        for (int j = 0; j < 4; ++j) {
            int gn   = n0 + nb + j * 16 + ln;
            float bv = bias[gn];
#pragma unroll
            for (int e = 0; e < 8; ++e) {
                int gm = m0 + mb + i * 16 + e + 8 * g;
                float val = acc[i][j][e] + bv;
                if (MODE == 0) {
                    unsigned short* O = (unsigned short*)Outv;   // [h][s][dk] bf16
                    int h = gn >> 6, d = gn & 63;
                    O[((size_t)h * SEQ + gm) * DK + d] = f2bf(val);
                } else {
                    float* O = (float*)Outv;                     // [s][dmodel] fp32
                    O[(size_t)gm * D_MODEL + gn] = val;
                }
            }
        }
}

// ---------------------------------------------------------------------------
// Flash attention: one (head, 128-query block) per 256-thread block.
// Q/K/V bf16 head-major [h][s][64]. Output ctx bf16 [s][1024].
// 8 waves, each owns 16 query rows; stream 64-key blocks with K/V register
// prefetch overlapping the scores/softmax/PV compute.
// ---------------------------------------------------------------------------
#define ATT_STRIDE 72   // 64 + 8 halfword pad

__global__ __launch_bounds__(256)
void flash_attn_kernel(const unsigned short* __restrict__ Qg,
                       const unsigned short* __restrict__ Kg,
                       const unsigned short* __restrict__ Vg,
                       unsigned short* __restrict__ Ctx) {
    __shared__ __align__(16) unsigned short sP[128 * ATT_STRIDE]; // P tiles / Q staging
    __shared__ __align__(16) unsigned short sK[64 * ATT_STRIDE];  // [n][d]
    __shared__ __align__(16) unsigned short sV[64 * ATT_STRIDE];  // transposed: [d][kn]

    const int tid  = threadIdx.x;
    const int qb   = blockIdx.x;          // 0..31
    const int h    = blockIdx.y;          // 0..15
    const int wv   = tid >> 5;            // wave 0..7 -> query rows wv*16..+15
    const int lane = tid & 31;
    const int g    = lane >> 4;
    const int ln   = lane & 15;

    const unsigned short* Qh = Qg + (size_t)h * SEQ * DK;
    const unsigned short* Kh = Kg + (size_t)h * SEQ * DK;
    const unsigned short* Vh = Vg + (size_t)h * SEQ * DK;
    const int s0 = qb * 128;

    // --- stage Q block (reuse sP), then keep fragments in registers ---
#pragma unroll
    for (int t = 0; t < 4; ++t) {
        int f8  = tid + t * 256;          // 1024 v8u's
        int row = f8 >> 3, c8 = f8 & 7;
        *(v8u*)(sP + row * ATT_STRIDE + c8 * 8) =
            *(const v8u*)(Qh + (size_t)(s0 + row) * DK + c8 * 8);
    }
    __syncthreads();
    v16bf aq0 = load_frag16(sP, wv * 16, ATT_STRIDE, 0);    // d 0..31
    v16bf aq1 = load_frag16(sP, wv * 16, ATT_STRIDE, 32);   // d 32..63
    __syncthreads();

    float mrow[8], lrow[8];
    v8f accO[4];
#pragma unroll
    for (int e = 0; e < 8; ++e) { mrow[e] = -3.0e38f; lrow[e] = 0.0f; }
#pragma unroll
    for (int t = 0; t < 4; ++t) accO[t] = zero8();

    const float scale = 0.125f;           // 1/sqrt(64)

    // K/V prefetch registers
    v8u kr0, kr1, vr0, vr1;
    const int krow = tid >> 3, kc8 = (tid & 7) * 8;   // K staging (2 rows apart by 32)
    const int vrow = tid >> 2, vc0 = (tid & 3) * 16;  // V staging (transpose)

    auto load_kv = [&](int kb) {
        kr0 = *(const v8u*)(Kh + (size_t)(kb * 64 + krow)      * DK + kc8);
        kr1 = *(const v8u*)(Kh + (size_t)(kb * 64 + krow + 32) * DK + kc8);
        vr0 = *(const v8u*)(Vh + (size_t)(kb * 64 + vrow) * DK + vc0);
        vr1 = *(const v8u*)(Vh + (size_t)(kb * 64 + vrow) * DK + vc0 + 8);
    };
    auto store_kv = [&]() {
        *(v8u*)(sK + krow * ATT_STRIDE + kc8)        = kr0;
        *(v8u*)(sK + (krow + 32) * ATT_STRIDE + kc8) = kr1;
#pragma unroll
        for (int i = 0; i < 8; ++i) sV[(vc0 + i) * ATT_STRIDE + vrow]     = vr0[i];
#pragma unroll
        for (int i = 0; i < 8; ++i) sV[(vc0 + 8 + i) * ATT_STRIDE + vrow] = vr1[i];
    };

    const int NKB = SEQ / 64;
    load_kv(0);

    for (int kb = 0; kb < NKB; ++kb) {
        store_kv();
        __syncthreads();
        if (kb + 1 < NKB) load_kv(kb + 1);   // overlap with compute

        // --- scores: S = Q * K^T, 4 n-tiles of 16 keys ---
        v8f st[4];
#pragma unroll
        for (int t = 0; t < 4; ++t) {
            v16bf b0 = load_frag16(sK, t * 16, ATT_STRIDE, 0);
            v16bf b1 = load_frag16(sK, t * 16, ATT_STRIDE, 32);
            v8f s = zero8();
            s = wmma_bf16(aq0, b0, s);
            s = wmma_bf16(aq1, b1, s);
#pragma unroll
            for (int e = 0; e < 8; ++e) s[e] *= scale;
            st[t] = s;
        }

        // --- online softmax (row stats per lane-group of 16) ---
#pragma unroll
        for (int e = 0; e < 8; ++e) {
            float rm = fmaxf(fmaxf(st[0][e], st[1][e]), fmaxf(st[2][e], st[3][e]));
#pragma unroll
            for (int msk = 1; msk < 16; msk <<= 1)
                rm = fmaxf(rm, __shfl_xor(rm, msk, 32));
            float mnew  = fmaxf(mrow[e], rm);
            float alpha = __expf(mrow[e] - mnew);
            mrow[e] = mnew;
            float rs = 0.0f;
#pragma unroll
            for (int t = 0; t < 4; ++t) {
                float p = __expf(st[t][e] - mnew);
                st[t][e] = p;
                rs += p;
            }
#pragma unroll
            for (int msk = 1; msk < 16; msk <<= 1)
                rs += __shfl_xor(rs, msk, 32);
            lrow[e] = lrow[e] * alpha + rs;
#pragma unroll
            for (int t = 0; t < 4; ++t) accO[t][e] *= alpha;
        }

        // --- P (D-layout) -> LDS [m][kn] so it can be re-read as an A frag ---
#pragma unroll
        for (int t = 0; t < 4; ++t)
#pragma unroll
            for (int e = 0; e < 8; ++e)
                sP[(wv * 16 + e + 8 * g) * ATT_STRIDE + t * 16 + ln] = f2bf(st[t][e]);
        __syncthreads();

        v16bf ap0 = load_frag16(sP, wv * 16, ATT_STRIDE, 0);    // kn 0..31
        v16bf ap1 = load_frag16(sP, wv * 16, ATT_STRIDE, 32);   // kn 32..63

        // --- accO += P * V  (4 d-tiles of 16) ---
#pragma unroll
        for (int t = 0; t < 4; ++t) {
            v16bf bv0 = load_frag16(sV, t * 16, ATT_STRIDE, 0);
            v16bf bv1 = load_frag16(sV, t * 16, ATT_STRIDE, 32);
            accO[t] = wmma_bf16(ap0, bv0, accO[t]);
            accO[t] = wmma_bf16(ap1, bv1, accO[t]);
        }
        __syncthreads();   // before next iteration overwrites sK/sV/sP
    }

    // --- epilogue: ctx = accO / l, bf16 row-major [s][1024] ---
#pragma unroll
    for (int t = 0; t < 4; ++t)
#pragma unroll
        for (int e = 0; e < 8; ++e) {
            int srow = s0 + wv * 16 + e + 8 * g;
            int col  = h * DK + t * 16 + ln;
            float val = accO[t][e] / lrow[e];
            Ctx[(size_t)srow * D_MODEL + col] = f2bf(val);
        }
}

// ---------------------------------------------------------------------------
// Host-side launcher
// ---------------------------------------------------------------------------
extern "C" void kernel_launch(void* const* d_in, const int* in_sizes, int n_in,
                              void* d_out, int out_size, void* d_ws, size_t ws_size,
                              hipStream_t stream) {
    const float* q  = (const float*)d_in[0];
    const float* k  = (const float*)d_in[1];
    const float* v  = (const float*)d_in[2];
    const float* Wq = (const float*)d_in[3];
    const float* bq = (const float*)d_in[4];
    const float* Wk = (const float*)d_in[5];
    const float* bk = (const float*)d_in[6];
    const float* Wv = (const float*)d_in[7];
    const float* bv = (const float*)d_in[8];
    const float* Wo = (const float*)d_in[9];
    const float* bo = (const float*)d_in[10];

    // workspace: Q, K, V (bf16 head-major) + ctx (bf16 row-major) = 32 MB
    const size_t elems = (size_t)SEQ * D_MODEL;
    unsigned short* wsQ = (unsigned short*)d_ws;
    unsigned short* wsK = wsQ + elems;
    unsigned short* wsV = wsK + elems;
    unsigned short* wsC = wsV + elems;

    dim3 gGemm(D_MODEL / 128, SEQ / 128);  // (8, 32)
    dim3 bGemm(256);

    gemm_bf16_kernel<0><<<gGemm, bGemm, 0, stream>>>(q, Wq, bq, wsQ);
    gemm_bf16_kernel<0><<<gGemm, bGemm, 0, stream>>>(k, Wk, bk, wsK);
    gemm_bf16_kernel<0><<<gGemm, bGemm, 0, stream>>>(v, Wv, bv, wsV);

    dim3 gAtt(SEQ / 128, NHEAD);           // (32, 16)
    flash_attn_kernel<<<gAtt, 256, 0, stream>>>(wsQ, wsK, wsV, wsC);

    gemm_bf16_kernel<1><<<gGemm, bGemm, 0, stream>>>(wsC, Wo, bo, (float*)d_out);
}
